// RNNGCNModel_45930380263836
// MI455X (gfx1250) — matrix operand order repaired and checked by reading, not compiled, not run on hardware
//
#include <hip/hip_runtime.h>
#include <hip/hip_bf16.h>
#include <math.h>

#define NNODES 50000
#define NEDGES 800000
#define DFEAT  128
#define NCLS   64

typedef __attribute__((ext_vector_type(16))) _Float16 v16h;
typedef __attribute__((ext_vector_type(8)))  _Float16 v8h;
typedef __attribute__((ext_vector_type(8)))  float    v8f;

// ---------------------------------------------------------------------------
// Pack kernels: f32 -> f16 conversions into WMMA-friendly layouts
// ---------------------------------------------------------------------------

// Acat[n][0:128] = topo_x[n], Acat[n][128:256] = hidden[0][n]   (f16)
__global__ __launch_bounds__(256) void pack_rnn_A_kernel(
    const float* __restrict__ topo, const float* __restrict__ hidden,
    _Float16* __restrict__ Acat)
{
    size_t i = (size_t)blockIdx.x * 256 + threadIdx.x;   // over N*128
    if (i >= (size_t)NNODES * DFEAT) return;
    size_t n = i >> 7;
    int    k = (int)(i & 127);
    Acat[n * 256 + k]       = (_Float16)topo[i];
    Acat[n * 256 + 128 + k] = (_Float16)hidden[i];
}

// Bt[o][k] = (k<128 ? W_ih[o][k] : W_hh[o][k-128]); bcomb = b_ih + b_hh
__global__ __launch_bounds__(256) void pack_rnn_B_kernel(
    const float* __restrict__ W_ih, const float* __restrict__ W_hh,
    const float* __restrict__ b_ih, const float* __restrict__ b_hh,
    _Float16* __restrict__ Bt, float* __restrict__ bcomb)
{
    int i = blockIdx.x * 256 + threadIdx.x;              // 128*256 entries
    if (i < 128 * 256) {
        int o = i >> 8, k = i & 255;
        float w = (k < 128) ? W_ih[o * 128 + k] : W_hh[o * 128 + (k - 128)];
        Bt[i] = (_Float16)w;
    }
    if (i < 128) bcomb[i] = b_ih[i] + b_hh[i];
}

// GCN weight W[K][F] row-major -> Bt[F][K] f16 (column-major for WMMA B frags)
__global__ __launch_bounds__(256) void pack_gcn_B_kernel(
    const float* __restrict__ W, _Float16* __restrict__ Bt, int K, int F)
{
    int i = blockIdx.x * 256 + threadIdx.x;
    if (i >= F * K) return;
    int o = i / K, k = i - o * K;
    Bt[i] = (_Float16)W[k * F + o];
}

// ---------------------------------------------------------------------------
// WMMA GEMM: C[M,NOUT] = A[M,KDIM] (f16) x Bt[NOUT,KDIM]^T (f16), f32 accum.
// One wave -> 16 rows x NOUT cols. Weights staged in LDS.
// Per k-step: clause-load ALL B fragments, then burst the WMMAs, so the
// matrix pipe sees one dscnt wait per NT WMMAs instead of one per WMMA.
// ---------------------------------------------------------------------------
template <int KDIM, int NOUT>
__global__ __launch_bounds__(256) void gemm_wmma_kernel(
    const _Float16* __restrict__ A,
    const _Float16* __restrict__ Bt,
    const float*    __restrict__ bias,     // nullable
    float*          __restrict__ Cf,       // nullable
    _Float16*       __restrict__ Ch,       // nullable
    int M, int doRelu)
{
    __shared__ __align__(16) _Float16 sB[NOUT * KDIM];
    const int tid = threadIdx.x;

    {   // cooperative 16B-chunk copy of Bt into LDS (ds_store_b128)
        const uint4* src = (const uint4*)Bt;
        uint4*       dst = (uint4*)sB;
        const int nchunks = (NOUT * KDIM) / 8;
        for (int i = tid; i < nchunks; i += 256) dst[i] = src[i];
    }
    __syncthreads();

    const int wave = tid >> 5;
    const int lane = tid & 31;
    const int row0 = (blockIdx.x * 8 + wave) * 16;
    if (row0 >= M) return;                     // wave-uniform

    const int r  = lane & 15;
    const int hi = lane >> 4;
    const _Float16* Arow = A + (size_t)(row0 + r) * KDIM;

    constexpr int NT = NOUT / 16;
    v8f acc[NT] = {};

#pragma unroll
    for (int k0 = 0; k0 < KDIM; k0 += 32) {
        // A 16x32 frag: halves [k0+8*hi .. +7] then [k0+16+8*hi .. +7]
        v8h alo = *(const v8h*)(Arow + k0 + 8 * hi);
        v8h ahi = *(const v8h*)(Arow + k0 + 16 + 8 * hi);
        v16h a = __builtin_shufflevector(alo, ahi,
                 0,1,2,3,4,5,6,7,8,9,10,11,12,13,14,15);

        // Batch-load all B fragments for this k-step (clause of ds_load_b128)
        v16h bfrag[NT];
#pragma unroll
        for (int t = 0; t < NT; ++t) {
            const _Float16* bp = &sB[(size_t)(t * 16 + r) * KDIM + k0 + 8 * hi];
            v8h blo = *(const v8h*)bp;
            v8h bhi = *(const v8h*)(bp + 16);
            bfrag[t] = __builtin_shufflevector(blo, bhi,
                       0,1,2,3,4,5,6,7,8,9,10,11,12,13,14,15);
        }

        // WMMA burst: NT back-to-back matrix ops sharing the A fragment
#pragma unroll
        for (int t = 0; t < NT; ++t) {
            acc[t] = __builtin_amdgcn_wmma_f32_16x16x32_f16(
                         /*neg_a=*/false, a, /*neg_b=*/false, bfrag[t],
                         /*c_mod=*/(short)0, acc[t],
                         /*reuse_a=*/false, /*reuse_b=*/false);
        }
    }

    // D layout: VGPR rr -> row rr + 8*hi, col = lane&15 (+ 16*t)
#pragma unroll
    for (int t = 0; t < NT; ++t) {
        const int col = t * 16 + r;
        const float bv = bias ? bias[col] : 0.0f;
#pragma unroll
        for (int rr = 0; rr < 8; ++rr) {
            const int row = row0 + rr + 8 * hi;
            if (row < M) {
                float v = acc[t][rr] + bv;
                if (doRelu) v = fmaxf(v, 0.0f);
                size_t idx = (size_t)row * NOUT + col;
                if (Cf) Cf[idx] = v;
                if (Ch) Ch[idx] = (_Float16)v;
            }
        }
    }
}

// ---------------------------------------------------------------------------
// SpMM scatter: one wave per edge, lanes stride the feature dim.
// Y[rows[e], :] += vals[e] * S[cols[e], :]   (global_atomic_add_f32, L2-bound)
// ---------------------------------------------------------------------------
__global__ __launch_bounds__(256) void spmm_scatter_kernel(
    const int* __restrict__ rows, const int* __restrict__ cols,
    const float* __restrict__ vals, const float* __restrict__ S,
    float* __restrict__ Y, int E, int F)
{
    const int e    = (int)((blockIdx.x * 256u + threadIdx.x) >> 5);
    const int lane = threadIdx.x & 31;
    if (e >= E) return;
    const int   dst = rows[e];
    const int   src = cols[e];
    const float v   = vals[e];
    const float* srow = S + (size_t)src * F;
    float*       drow = Y + (size_t)dst * F;
    for (int f = lane; f < F; f += 32)
        atomicAdd(&drow[f], v * srow[f]);
}

// x = relu(Y + bias[c]); write f32 and/or f16 copies
__global__ __launch_bounds__(256) void bias_act_kernel(
    const float* __restrict__ Y, const float* __restrict__ bias,
    float* __restrict__ Xf, _Float16* __restrict__ Xh,
    size_t total, int cmask)
{
    size_t i = (size_t)blockIdx.x * 256 + threadIdx.x;
    if (i >= total) return;
    int   c = (int)(i & (size_t)cmask);
    float v = fmaxf(Y[i] + bias[c], 0.0f);
    if (Xf) Xf[i] = v;
    if (Xh) Xh[i] = (_Float16)v;
}

// Per-column logsumexp over the node dim: one block per class channel
__global__ __launch_bounds__(256) void col_lse_kernel(
    const float* __restrict__ X, int Nn, int C, float* __restrict__ lse)
{
    __shared__ float red[256];
    const int c = blockIdx.x;
    float m = -INFINITY;
    for (int n = threadIdx.x; n < Nn; n += 256)
        m = fmaxf(m, X[(size_t)n * C + c]);
    red[threadIdx.x] = m; __syncthreads();
    for (int s = 128; s > 0; s >>= 1) {
        if (threadIdx.x < s)
            red[threadIdx.x] = fmaxf(red[threadIdx.x], red[threadIdx.x + s]);
        __syncthreads();
    }
    m = red[0]; __syncthreads();
    float acc = 0.0f;
    for (int n = threadIdx.x; n < Nn; n += 256)
        acc += expf(X[(size_t)n * C + c] - m);
    red[threadIdx.x] = acc; __syncthreads();
    for (int s = 128; s > 0; s >>= 1) {
        if (threadIdx.x < s) red[threadIdx.x] += red[threadIdx.x + s];
        __syncthreads();
    }
    if (threadIdx.x == 0) lse[c] = m + logf(red[0]);
}

__global__ __launch_bounds__(256) void softmax_out_kernel(
    const float* __restrict__ X, const float* __restrict__ lse,
    float* __restrict__ out, size_t total, int cmask)
{
    size_t i = (size_t)blockIdx.x * 256 + threadIdx.x;
    if (i >= total) return;
    out[i] = X[i] - lse[i & (size_t)cmask];
}

// ---------------------------------------------------------------------------
// Launcher
// ---------------------------------------------------------------------------
extern "C" void kernel_launch(void* const* d_in, const int* in_sizes, int n_in,
                              void* d_out, int out_size, void* d_ws, size_t ws_size,
                              hipStream_t stream)
{
    (void)in_sizes; (void)n_in; (void)out_size; (void)ws_size;

    // inputs (setup_inputs order); feat_x (d_in[0]) unused because _comb=='topo'
    const float* topo_x   = (const float*)d_in[1];
    const int*   adj_rows = (const int*)  d_in[2];
    const int*   adj_cols = (const int*)  d_in[3];
    const float* adj_vals = (const float*)d_in[4];
    const float* hidden   = (const float*)d_in[5];
    const float* W_ih     = (const float*)d_in[6];
    const float* W_hh     = (const float*)d_in[7];
    const float* b_ih     = (const float*)d_in[8];
    const float* b_hh     = (const float*)d_in[9];
    const float* gw0      = (const float*)d_in[10];
    const float* gb0      = (const float*)d_in[11];
    const float* gw1      = (const float*)d_in[12];
    const float* gb1      = (const float*)d_in[13];
    const float* gw2      = (const float*)d_in[14];
    const float* gb2      = (const float*)d_in[15];

    float* out_ls = (float*)d_out;                              // [N, C]
    float* out_h  = (float*)d_out + (size_t)NNODES * NCLS;      // [1, N, D]

    // workspace carve-up (256B aligned)
    char*  ws  = (char*)d_ws;
    size_t off = 0;
    auto wsAlloc = [&](size_t bytes) -> void* {
        void* p = ws + off;
        off = (off + bytes + 255) & ~(size_t)255;
        return p;
    };
    _Float16* Acat    = (_Float16*)wsAlloc((size_t)NNODES * 256 * 2);   // [N,256] f16
    _Float16* xh      = (_Float16*)wsAlloc((size_t)NNODES * DFEAT * 2); // [N,128] f16
    float*    support = (float*)   wsAlloc((size_t)NNODES * DFEAT * 4); // [N,128] f32
    float*    ybuf    = (float*)   wsAlloc((size_t)NNODES * DFEAT * 4); // [N,128] f32
    _Float16* BtR     = (_Float16*)wsAlloc((size_t)128 * 256 * 2);
    _Float16* Bt0     = (_Float16*)wsAlloc((size_t)128 * 128 * 2);
    _Float16* Bt1     = (_Float16*)wsAlloc((size_t)128 * 128 * 2);
    _Float16* Bt2     = (_Float16*)wsAlloc((size_t)64  * 128 * 2);
    float*    bcomb   = (float*)   wsAlloc(128 * 4);
    float*    lse     = (float*)   wsAlloc(NCLS * 4);

    const int packA_blocks = (int)(((size_t)NNODES * DFEAT + 255) / 256);
    const int gemm_blocks  = (NNODES / 16 + 7) / 8;             // 3125 tiles -> 391
    const int spmm_blocks  = (int)(((size_t)NEDGES * 32 + 255) / 256);
    const int ea128_blocks = (int)(((size_t)NNODES * DFEAT + 255) / 256);
    const int ea64_blocks  = (int)(((size_t)NNODES * NCLS + 255) / 256);

    // ---- pack everything to f16 once ------------------------------------
    pack_rnn_A_kernel<<<packA_blocks, 256, 0, stream>>>(topo_x, hidden, Acat);
    pack_rnn_B_kernel<<<128, 256, 0, stream>>>(W_ih, W_hh, b_ih, b_hh, BtR, bcomb);
    pack_gcn_B_kernel<<<64, 256, 0, stream>>>(gw0, Bt0, DFEAT, DFEAT);
    pack_gcn_B_kernel<<<64, 256, 0, stream>>>(gw1, Bt1, DFEAT, DFEAT);
    pack_gcn_B_kernel<<<32, 256, 0, stream>>>(gw2, Bt2, DFEAT, NCLS);

    // ---- RNN step: h1 = relu([topo|h0] x [W_ih^T;W_hh^T] + b_ih + b_hh) -
    // writes f32 new_hidden output and f16 copy for the GCN stack
    gemm_wmma_kernel<256, 128><<<gemm_blocks, 256, 0, stream>>>(
        Acat, BtR, bcomb, out_h, xh, NNODES, 1);

    // ---- GCN layer 0: support = x W0 ; x = relu(A support + b0) ---------
    gemm_wmma_kernel<128, 128><<<gemm_blocks, 256, 0, stream>>>(
        xh, Bt0, nullptr, support, nullptr, NNODES, 0);
    hipMemsetAsync(ybuf, 0, (size_t)NNODES * DFEAT * 4, stream);
    spmm_scatter_kernel<<<spmm_blocks, 256, 0, stream>>>(
        adj_rows, adj_cols, adj_vals, support, ybuf, NEDGES, DFEAT);
    bias_act_kernel<<<ea128_blocks, 256, 0, stream>>>(
        ybuf, gb0, nullptr, xh, (size_t)NNODES * DFEAT, DFEAT - 1);

    // ---- GCN layer 1 -----------------------------------------------------
    gemm_wmma_kernel<128, 128><<<gemm_blocks, 256, 0, stream>>>(
        xh, Bt1, nullptr, support, nullptr, NNODES, 0);
    hipMemsetAsync(ybuf, 0, (size_t)NNODES * DFEAT * 4, stream);
    spmm_scatter_kernel<<<spmm_blocks, 256, 0, stream>>>(
        adj_rows, adj_cols, adj_vals, support, ybuf, NEDGES, DFEAT);
    bias_act_kernel<<<ea128_blocks, 256, 0, stream>>>(
        ybuf, gb1, nullptr, xh, (size_t)NNODES * DFEAT, DFEAT - 1);

    // ---- GCN layer 2 (128 -> 64) ----------------------------------------
    gemm_wmma_kernel<128, 64><<<gemm_blocks, 256, 0, stream>>>(
        xh, Bt2, nullptr, support, nullptr, NNODES, 0);
    hipMemsetAsync(ybuf, 0, (size_t)NNODES * NCLS * 4, stream);
    spmm_scatter_kernel<<<spmm_blocks, 256, 0, stream>>>(
        adj_rows, adj_cols, adj_vals, support, ybuf, NEDGES, NCLS);
    bias_act_kernel<<<ea64_blocks, 256, 0, stream>>>(
        ybuf, gb2, ybuf /*in-place f32*/, nullptr, (size_t)NNODES * NCLS, NCLS - 1);

    // ---- log_softmax over the node dimension (per-column LSE) -----------
    col_lse_kernel<<<NCLS, 256, 0, stream>>>(ybuf, NNODES, NCLS, lse);
    softmax_out_kernel<<<ea64_blocks, 256, 0, stream>>>(
        ybuf, lse, out_ls, (size_t)NNODES * NCLS, NCLS - 1);
}
